// GGMLLayer_22162031247686
// MI455X (gfx1250) — compile-verified, matrix operand order checked by reading, not run on hardware
//
#include <hip/hip_runtime.h>

// Q8_0 dequant + GEMM: out[128,11008] = x[128,4096] @ W^T + bias
// HBM-bound on the 180MB int32 weight stream (~8us floor at 23.3 TB/s).
// Weight path: HBM --TDM--> LDS raw -> cooperative dequant -> LDS f16 -> WMMA.
// Activation path: one f32->f16 prepass into d_ws (L2-resident), b128 loads.

#define IN_F    4096
#define OUT_F   11008
#define NTILE   16
#define BPR     128                   // 32-blocks per weight row
#define CHUNK_K 256                   // int32 k-elements per TDM chunk per row
#define NCHUNKS (IN_F / CHUNK_K)      // 16
#define STEP_K  64                    // k per inner step (2 WMMAs)
#define NSTEPS  (CHUNK_K / STEP_K)    // 4

typedef __attribute__((ext_vector_type(16))) _Float16 v16h;
typedef __attribute__((ext_vector_type(8)))  _Float16 v8h;
typedef __attribute__((ext_vector_type(4)))  _Float16 v4h;
typedef __attribute__((ext_vector_type(8)))  float    v8f;
typedef __attribute__((ext_vector_type(4)))  float    v4f;
typedef __attribute__((ext_vector_type(4)))  unsigned int u32x4;
typedef __attribute__((ext_vector_type(8)))  int      i32x8;
typedef __attribute__((ext_vector_type(4)))  int      i32x4;

// f16 tile: column stride 72 halfs = 144B (16B aligned, bank-skewed so 16
// lanes hitting 16 columns with ds_load_b128 are conflict-free).
#define FSTRIDE 72
#define FBUFH   (NTILE * FSTRIDE)     // 1152 halfs
#define RAWH    (NTILE * CHUNK_K)     // 4096 ints = 16KB per raw buffer

#define CAT16(a, b) __builtin_shufflevector(a, b, \
        0, 1, 2, 3, 4, 5, 6, 7, 8, 9, 10, 11, 12, 13, 14, 15)

// ---- Tensor Data Mover: 2D tile load, 16 rows x CHUNK_K int32, row stride
// IN_F int32.  D# packed per CDNA5 ISA 8.3/8.4 (count=1, type=2, dsize=4B).
// clang-23 toolchain: 6-arg builtin (g0 u32x4, g1 i32x8, g2 i32x4, g3 i32x4,
// extra i32x8, cpol i32); trailing groups zero-filled for a 2D tensor.
__device__ __forceinline__ void tdm_load(const int* gsrc, unsigned ldsOff,
                                         int rowsLeft, int kLeft)
{
    unsigned long long ga = (unsigned long long)(const void*)gsrc;
    u32x4 g0;
    g0[0] = 1u;                                            // count=1, load, no gather
    g0[1] = ldsOff;                                        // lds_addr (bytes)
    g0[2] = (unsigned)ga;                                  // global_addr[31:0]
    g0[3] = (unsigned)((ga >> 32) & 0x01FFFFFFu) | (2u << 30); // addr[56:32] | type=2
    i32x8 g1;
    g1[0] = 2 << 16;                                       // data_size=4B; no pad/iter/mcast
    g1[1] = (kLeft & 0xFFFF) << 16;                        // tensor_dim0[15:0]
    g1[2] = ((kLeft >> 16) & 0xFFFF) | ((rowsLeft & 0xFFFF) << 16); // dim0_hi | dim1_lo
    g1[3] = ((rowsLeft >> 16) & 0xFFFF) | (CHUNK_K << 16); // dim1_hi | tile_dim0
    g1[4] = NTILE;                                         // tile_dim1=16, tile_dim2=0
    g1[5] = IN_F;                                          // tensor_dim0_stride[31:0]
    g1[6] = 0;                                             // stride0_hi | dim1_stride_lo
    g1[7] = 0;
    i32x4 gz4 = {0, 0, 0, 0};                              // groups 2/3 unused (2D)
    i32x8 gz8 = {0, 0, 0, 0, 0, 0, 0, 0};
    __builtin_amdgcn_tensor_load_to_lds(g0, g1, gz4, gz4, gz8, 0);
}

// ---- prepass: x f32 -> f16 into workspace (524288 elems, 1MB out)
__global__ __launch_bounds__(256) void xcvt_kernel(const float* __restrict__ x,
                                                   _Float16* __restrict__ xh)
{
    const int i = (blockIdx.x * 256 + threadIdx.x) * 4;
    v4f f = *(const v4f*)(x + i);
    v4h h;
    h[0] = (_Float16)f[0];
    h[1] = (_Float16)f[1];
    h[2] = (_Float16)f[2];
    h[3] = (_Float16)f[3];
    *(v4h*)(xh + i) = h;
}

__global__ __launch_bounds__(256) void q8_gemm_wmma_tdm_kernel(
    const _Float16* __restrict__ xh,     // [128, 4096] f16 (prepass)
    const float*    __restrict__ scales, // [1409024]
    const int*      __restrict__ qw,     // [11008, 4096] int32 (emulated int8)
    const float*    __restrict__ bias,   // [11008]
    float*          __restrict__ out)    // [128, 11008]
{
    __shared__ int      sRaw[2 * RAWH];   // 32KB raw int32 (TDM destination)
    __shared__ _Float16 sB[2 * FBUFH];    // 4.5KB dequantized f16 tiles

    const int tid  = threadIdx.x;
    const int lane = tid & 31;
    const int wave = tid >> 5;              // 0..7 -> M tile of 16 rows
    const int n0   = blockIdx.x * NTILE;

    // uniform (SGPR) condition -> scalar branch; waves 1..7 skip TDM entirely
    const bool tdmWave = (__builtin_amdgcn_readfirstlane((int)tid) >> 5) == 0;

    // ---- cooperative dequant decomposition: 256 thr cover 16 cols x 64 k
    const int scol     = tid >> 4;          // weight column 0..15
    const int skq      = (tid & 15) * 4;    // 4 consecutive k per thread
    const int sBlkBase = (n0 + scol) * BPR;

    // ---- WMMA fragment indices
    const int ccol = lane & 15;             // N within tile
    const int koff = (lane < 16) ? 0 : 8;   // 16-bit fragment lane split
    const _Float16* xrow = xh + (long)(wave * NTILE + ccol) * IN_F;

    const unsigned rawLds[2] = {
        (unsigned)(size_t)(void*)&sRaw[0],
        (unsigned)(size_t)(void*)&sRaw[RAWH]
    };

    v8f acc = {};

    const int* gbase = qw + (long)n0 * IN_F;

    if (tdmWave)
        tdm_load(gbase, rawLds[0], OUT_F - n0, IN_F);      // chunk 0

    for (int c = 0; c < NCHUNKS; ++c) {
        if (tdmWave)
            __builtin_amdgcn_s_wait_tensorcnt((short)0);   // chunk c resident
        __syncthreads();                                    // publish raw chunk c

        if (tdmWave && (c + 1 < NCHUNKS))                   // overlap DMA c+1
            tdm_load(gbase + (c + 1) * CHUNK_K, rawLds[(c + 1) & 1],
                     OUT_F - n0, IN_F - (c + 1) * CHUNK_K);

        const int* rawCur = &sRaw[(c & 1) * RAWH];

        for (int s = 0; s < NSTEPS; ++s) {
            const int kb = s * STEP_K;                      // k within chunk

            // ---- dequant 64-k f16 tile into LDS (once per block, not per wave)
            {
                i32x4 q = *(const i32x4*)(rawCur + scol * CHUNK_K + kb + skq);
                float sc = scales[sBlkBase + ((c * CHUNK_K + kb + skq) >> 5)];
                v4h hv;
                hv[0] = (_Float16)((float)(q[0] - 128) * sc);
                hv[1] = (_Float16)((float)(q[1] - 128) * sc);
                hv[2] = (_Float16)((float)(q[2] - 128) * sc);
                hv[3] = (_Float16)((float)(q[3] - 128) * sc);
                *(v4h*)(&sB[(s & 1) * FBUFH + scol * FSTRIDE + skq]) = hv;
            }
            __syncthreads();                                // tile visible

            // ---- consume: two v_wmma_f32_16x16x32_f16 (k and k+32)
            const _Float16* bp = &sB[(s & 1) * FBUFH + ccol * FSTRIDE];
            const _Float16* ap = xrow + c * CHUNK_K + kb;
#pragma unroll
            for (int h = 0; h < 2; ++h) {
                const int k2 = h * 32 + koff;
                v8h a0 = *(const v8h*)(ap + k2);
                v8h a1 = *(const v8h*)(ap + k2 + 16);
                v8h b0 = *(const v8h*)(bp + k2);
                v8h b1 = *(const v8h*)(bp + k2 + 16);
                v16h A = CAT16(a0, a1);
                v16h B = CAT16(b0, b1);
                acc = __builtin_amdgcn_wmma_f32_16x16x32_f16(
                          false, A, false, B, (short)0, acc, false, false);
            }
            // no trailing barrier needed: next step writes the other f16
            // buffer; reuse of this one is fenced by the next step's barrier
        }
    }

    // ---- epilogue: bias + store per 16x16 f32 C layout
    const int   n     = n0 + ccol;
    const float bv    = bias[n];
    const int   mBase = wave * NTILE + ((lane < 16) ? 0 : 8);
#pragma unroll
    for (int r = 0; r < 8; ++r) {
        out[(long)(mBase + r) * OUT_F + n] = acc[r] + bv;
    }
}

extern "C" void kernel_launch(void* const* d_in, const int* in_sizes, int n_in,
                              void* d_out, int out_size, void* d_ws, size_t ws_size,
                              hipStream_t stream) {
    const float* x      = (const float*)d_in[0];
    const float* scales = (const float*)d_in[1];
    const int*   qw     = (const int*)d_in[2];
    const float* bias   = (const float*)d_in[3];
    float*       out    = (float*)d_out;
    _Float16*    xh     = (_Float16*)d_ws;      // 1MB f16 activations

    xcvt_kernel<<<dim3((128 * IN_F) / (256 * 4)), dim3(256), 0, stream>>>(x, xh);
    q8_gemm_wmma_tdm_kernel<<<dim3(OUT_F / NTILE), dim3(256), 0, stream>>>(
        xh, scales, qw, bias, out);
}
